// Geo_HR_SNN_8890582302805
// MI455X (gfx1250) — matrix operand order repaired and checked by reading, not compiled, not run on hardware
//
#include <hip/hip_runtime.h>
#include <cstdint>

#define BB  1024
#define TT  320
#define HID 32

typedef __attribute__((ext_vector_type(8)))  float  v8f;
typedef __attribute__((ext_vector_type(16))) __bf16 v16bf;

union Frag { uint4 q[2]; v16bf v; };

__device__ __forceinline__ unsigned short f2bf(float f) {
  unsigned int u = __float_as_uint(f);
  u += 0x7FFFu + ((u >> 16) & 1u);            // round-to-nearest-even
  return (unsigned short)(u >> 16);
}
__device__ __forceinline__ float bf2f(unsigned short h) {
  return __uint_as_float(((unsigned int)h) << 16);
}

// ---------------------------------------------------------------- prep ----
struct PrepArgs {
  const float *ecw, *ecb, *es1, *eb1, *em1, *ev1, *edw, *edb, *es2, *eb2, *em2, *ev2;
  const float *gcw, *gcb, *gs1, *gb1, *gm1, *gv1, *gdw, *gdb, *gs2, *gb2, *gm2, *gv2;
  const float *fc1w;
  unsigned short *wtE, *wtG, *f1T;
  float *sco;  // [scaleE 32 | offE 32 | scaleG 32 | offG 32]
};

__global__ __launch_bounds__(256) void prep_kernel(PrepArgs p) {
  int gid = blockIdx.x * 256 + threadIdx.x;
  if (gid < 65536) {                      // WeffT_eeg[h][j], j = k*64 + chin
    int h = gid >> 11, j = gid & 2047;
    int k = j >> 6, ch = j & 63;
    float s = 0.f;
#pragma unroll
    for (int f = 0; f < 8; ++f) {
      int c = ch * 8 + f;
      float a1 = p.es1[c] * rsqrtf(p.ev1[c] + 1e-5f);
      s += p.ecw[k * 512 + c] * a1 * p.edw[c * 32 + h];
    }
    p.wtE[h * 2048 + j] = f2bf(s);
  } else if (gid < 65536 + 32768) {       // WeffT_geo[h][j], j = k*32 + chin (chin>=18 -> 0)
    int i = gid - 65536;
    int h = i >> 10, j = i & 1023;
    int k = j >> 5, ch = j & 31;
    float s = 0.f;
    if (ch < 18) {
#pragma unroll
      for (int f = 0; f < 8; ++f) {
        int c = ch * 8 + f;
        float a1 = p.gs1[c] * rsqrtf(p.gv1[c] + 1e-5f);
        s += p.gcw[k * 144 + c] * a1 * p.gdw[c * 32 + h];
      }
    }
    p.wtG[h * 1024 + j] = f2bf(s);
  } else if (gid < 65536 + 32768 + 40960) {  // fc1wT[col][k]
    int i = gid - 98304;
    int col = i / 320, k = i - col * 320;
    p.f1T[col * 320 + k] = f2bf(p.fc1w[k * 128 + col]);
  } else if (gid < 65536 + 32768 + 40960 + 64) {  // folded BN2 scale/offset
    int i = gid - 139264;
    int h = i & 31;
    if (i < 32) {
      float a2 = p.es2[h] * rsqrtf(p.ev2[h] + 1e-5f);
      float C = p.edb[h];
      for (int c = 0; c < 512; ++c) {
        float a1 = p.es1[c] * rsqrtf(p.ev1[c] + 1e-5f);
        float be = p.eb1[c] - p.em1[c] * a1;
        C += (a1 * p.ecb[c] + be) * p.edw[c * 32 + h];
      }
      p.sco[h]      = a2;
      p.sco[32 + h] = a2 * C + (p.eb2[h] - p.em2[h] * a2);
    } else {
      float a2 = p.gs2[h] * rsqrtf(p.gv2[h] + 1e-5f);
      float C = p.gdb[h];
      for (int c = 0; c < 144; ++c) {
        float a1 = p.gs1[c] * rsqrtf(p.gv1[c] + 1e-5f);
        float be = p.gb1[c] - p.gm1[c] * a1;
        C += (a1 * p.gcb[c] + be) * p.gdw[c * 32 + h];
      }
      p.sco[64 + h] = a2;
      p.sco[96 + h] = a2 * C + (p.gb2[h] - p.gm2[h] * a2);
    }
  }
}

// ------------------------------------------------------------- encoder ----
// One block = one (b, 64-wide t chunk). 8 waves = 4 M-tiles(time) x 2 N-tiles(hid).
// Conv+BN+dense+BN fused into one bf16 WMMA GEMM over j = k*CHP + chin.
template <int CHP, int CHIN, int NSTEPS, bool IS_GEO>
__global__ __launch_bounds__(256) void encoder_kernel(
    const float* __restrict__ x, const unsigned short* __restrict__ WT,
    const float* __restrict__ sco, unsigned short* __restrict__ enc) {
  __shared__ unsigned short xs[96 * CHP];
  const int b  = blockIdx.x / 5;
  const int t0 = (blockIdx.x % 5) * 64;
  for (int i = threadIdx.x; i < 96 * CHP; i += 256) {
    int r = i / CHP, c = i - r * CHP;
    int t = t0 - 16 + r;                 // SAME pad: y[t] = sum_k x[t+k-15]*w[k]
    float v = 0.f;
    if (t >= 0 && t < TT && c < CHIN) v = x[(b * TT + t) * CHIN + c];
    xs[i] = f2bf(v);
  }
  __syncthreads();

  const int lane  = threadIdx.x & 31;
  const int wave  = threadIdx.x >> 5;
  const int mtile = wave >> 1;
  const int ntile = wave & 1;
  const int lm    = lane & 15;
  const bool hi   = lane >= 16;

  v8f acc = {0.f, 0.f, 0.f, 0.f, 0.f, 0.f, 0.f, 0.f};
  const unsigned short* brow = WT + (ntile * 16 + lm) * (NSTEPS * 32) + (hi ? 8 : 0);
  const int arow0 = mtile * 16 + lm + 1;  // LDS row = t - (t0-16) = m + k + 1

  for (int s = 0; s < NSTEPS; ++s) {
    int k, chb;
    if (CHP == 64) { k = s >> 1; chb = (s & 1) << 5; } else { k = s; chb = 0; }
    // A fragment: lanes 0-15 hold Klocal {0..7,16..23}, lanes 16-31 {8..15,24..31}
    const unsigned short* ap = &xs[(arow0 + k) * CHP + chb + (hi ? 8 : 0)];
    Frag a, bb;
    a.q[0] = *(const uint4*)(ap);
    a.q[1] = *(const uint4*)(ap + 16);
    // B fragment: lane = column h, Klocal {0..7,16..23} / {8..15,24..31} like A
    const unsigned short* bp = brow + s * 32;
    bb.q[0] = *(const uint4*)(bp);
    bb.q[1] = *(const uint4*)(bp + 16);
    __builtin_prefetch((const void*)(bp + 128), 0, 1);
    acc = __builtin_amdgcn_wmma_f32_16x16x32_bf16(false, a.v, false, bb.v,
                                                  (short)0, acc, false, false);
  }

  const int   hcol = ntile * 16 + lm;
  const float sc = sco[hcol];
  const float of = sco[32 + hcol];
  const int   tb = t0 + mtile * 16 + (hi ? 8 : 0);
#pragma unroll
  for (int r = 0; r < 8; ++r) {
    float v = fmaf(sc, acc[r], of);
    if (IS_GEO) v = 1.f / (1.f + __expf(-v));
    else        v = fmaxf(v, 0.f);
    enc[(b * TT + (tb + r)) * HID + hcol] = f2bf(v);
  }
}

// ---------------------------------------------------------------- scan ----
// Block = 8 batches x 32 hid. Per 32-step window, stage both streams into LDS
// with GLOBAL_LOAD_ASYNC_TO_LDS_B128 (ASYNCcnt), then run the LIF recurrence
// from LDS. Each window per stream = 8 contiguous 2KB rows -> pure B128 bursts.
__global__ __launch_bounds__(256) void scan_kernel(
    const unsigned short* __restrict__ enc, const unsigned short* __restrict__ mod,
    const float* __restrict__ gamma, unsigned short* __restrict__ dp) {
  __shared__ unsigned short ebuf[8 * 32 * 32];   // [lb][tloc][h]  16KB
  __shared__ unsigned short gbuf[8 * 32 * 32];   // 16KB
  const int tid = threadIdx.x;
  const int b0  = blockIdx.x * 8;
  const int lb  = tid >> 5, h = tid & 31;
  const float g = gamma[h];
  const unsigned ebase = (unsigned)(uintptr_t)ebuf;
  const unsigned gbase = (unsigned)(uintptr_t)gbuf;
  const unsigned long long esrc0 = (unsigned long long)(uintptr_t)enc;
  const unsigned long long gsrc0 = (unsigned long long)(uintptr_t)mod;

  float m1 = 0.f, m2 = 0.f, m3 = 0.f, ma = 0.f, eta = 0.f, mli = 0.f, prev = 0.f;
  for (int w = 0; w < 10; ++w) {
    // ---- async stage: 16KB per stream, 4 B128 per thread per stream ----
#pragma unroll
    for (int j = 0; j < 4; ++j) {
      int flat = tid + j * 256;                       // 16-byte units, 0..1023
      int sb   = flat >> 7;                           // local batch (2KB each)
      unsigned long long rowb =
          ((unsigned long long)(b0 + sb) * TT + (unsigned long long)w * 32) * 64ull
          + (unsigned long long)((flat & 127) << 4);  // byte offset in stream
      unsigned      ldst = (unsigned)(flat << 4);
      unsigned long long ea = esrc0 + rowb;
      unsigned long long ga = gsrc0 + rowb;
      unsigned ed = ebase + ldst;
      unsigned gd = gbase + ldst;
      asm volatile("global_load_async_to_lds_b128 %0, %1, off"
                   :: "v"(ed), "v"(ea) : "memory");
      asm volatile("global_load_async_to_lds_b128 %0, %1, off"
                   :: "v"(gd), "v"(ga) : "memory");
    }
    asm volatile("s_wait_asynccnt 0x0" ::: "memory");
    __syncthreads();

    // ---- LIF recurrence over this 32-step window ----
    float a = 0.f;
#pragma unroll 4
    for (int i = 0; i < 32; ++i) {
      float ce = bf2f(ebuf[(lb * 32 + i) * 32 + h]);
      float cg = bf2f(gbuf[(lb * 32 + i) * 32 + h]);
      m1 = fmaf(0.9f, m1, ce); float s1 = (m1 >= 0.7f) ? 1.f : 0.f; m1 *= (1.f - s1);
      m2 = fmaf(0.8f, m2, ce); float s2 = (m2 >= 0.5f) ? 1.f : 0.f; m2 *= (1.f - s2);
      m3 = fmaf(0.6f, m3, ce); float s3 = (m3 >= 0.3f) ? 1.f : 0.f; m3 *= (1.f - s3);
      float hr = (s1 + s2 + s3) * (1.f / 3.f);
      eta = 0.36f * eta + 0.64f * prev;
      float theta = 0.5f + 1.8f * eta - g * cg;
      ma = fmaf(0.8f, ma, hr);
      float sa = (ma >= theta) ? 1.f : 0.f;
      ma *= (1.f - sa);
      mli = fmaf(0.9f, mli, sa);
      prev = sa;
      a += (i >= 16) ? mli : -mli;
    }
    __syncthreads();   // buffers must drain before next window's async writes
    dp[(size_t)(b0 + lb) * 320 + h * 10 + w] = f2bf(a * (1.f / 16.f));
  }
}

// ---------------------------------------------------------------- head ----
// block = 16 rows of B; 8 waves = 8 N-tiles over 128 fc1 cols; WMMA fc1 + ELU,
// then scalar fc2 (128 -> 4) from LDS (rows padded to 132 to avoid bank conflicts).
__global__ __launch_bounds__(256) void head_kernel(
    const unsigned short* __restrict__ dp, const unsigned short* __restrict__ f1T,
    const float* __restrict__ fc1b, const float* __restrict__ fc2w,
    const float* __restrict__ fc2b, float* __restrict__ out) {
  __shared__ float hbuf[16 * 132];
  const int b0   = blockIdx.x * 16;
  const int lane = threadIdx.x & 31;
  const int wave = threadIdx.x >> 5;
  const int lm   = lane & 15;
  const bool hi  = lane >= 16;
  const int col  = wave * 16 + lm;

  v8f acc = {0.f, 0.f, 0.f, 0.f, 0.f, 0.f, 0.f, 0.f};
  const unsigned short* arow = dp + (size_t)(b0 + lm) * 320 + (hi ? 8 : 0);
  const unsigned short* brow = f1T + (size_t)col * 320 + (hi ? 8 : 0);
  for (int s = 0; s < 10; ++s) {
    Frag a, bb;
    const unsigned short* ap = arow + s * 32;
    a.q[0] = *(const uint4*)(ap);
    a.q[1] = *(const uint4*)(ap + 16);
    const unsigned short* bp = brow + s * 32;
    bb.q[0] = *(const uint4*)(bp);
    bb.q[1] = *(const uint4*)(bp + 16);
    acc = __builtin_amdgcn_wmma_f32_16x16x32_bf16(false, a.v, false, bb.v,
                                                  (short)0, acc, false, false);
  }
  float bias = fc1b[col];
#pragma unroll
  for (int r = 0; r < 8; ++r) {
    float v = acc[r] + bias;
    v = (v > 0.f) ? v : (__expf(v) - 1.f);           // ELU
    hbuf[(r + (hi ? 8 : 0)) * 132 + col] = v;
  }
  __syncthreads();
  if (threadIdx.x < 64) {
    int row = threadIdx.x >> 2, o = threadIdx.x & 3;
    float s = fc2b[o];
    for (int c = 0; c < 128; ++c) s = fmaf(hbuf[row * 132 + c], fc2w[c * 4 + o], s);
    out[(b0 + row) * 4 + o] = s;
  }
}

// -------------------------------------------------------------- launch ----
#define OFF_WTE 0ull
#define OFF_WTG 131072ull
#define OFF_F1T 196608ull
#define OFF_SCO 278528ull
#define OFF_ENC 279040ull
#define OFF_MOD (279040ull + 20971520ull)
#define OFF_DP  (279040ull + 2ull * 20971520ull)

extern "C" void kernel_launch(void* const* d_in, const int* in_sizes, int n_in,
                              void* d_out, int out_size, void* d_ws, size_t ws_size,
                              hipStream_t stream) {
  (void)in_sizes; (void)n_in; (void)out_size; (void)ws_size;
  const float* x_eeg = (const float*)d_in[0];
  const float* x_geo = (const float*)d_in[1];
  unsigned char* ws = (unsigned char*)d_ws;
  unsigned short* wtE  = (unsigned short*)(ws + OFF_WTE);
  unsigned short* wtG  = (unsigned short*)(ws + OFF_WTG);
  unsigned short* f1T  = (unsigned short*)(ws + OFF_F1T);
  float*          sco  = (float*)(ws + OFF_SCO);
  unsigned short* encE = (unsigned short*)(ws + OFF_ENC);
  unsigned short* modG = (unsigned short*)(ws + OFF_MOD);
  unsigned short* dpb  = (unsigned short*)(ws + OFF_DP);

  PrepArgs p;
  p.ecw = (const float*)d_in[2];  p.ecb = (const float*)d_in[3];
  p.es1 = (const float*)d_in[4];  p.eb1 = (const float*)d_in[5];
  p.em1 = (const float*)d_in[6];  p.ev1 = (const float*)d_in[7];
  p.edw = (const float*)d_in[8];  p.edb = (const float*)d_in[9];
  p.es2 = (const float*)d_in[10]; p.eb2 = (const float*)d_in[11];
  p.em2 = (const float*)d_in[12]; p.ev2 = (const float*)d_in[13];
  p.gcw = (const float*)d_in[14]; p.gcb = (const float*)d_in[15];
  p.gs1 = (const float*)d_in[16]; p.gb1 = (const float*)d_in[17];
  p.gm1 = (const float*)d_in[18]; p.gv1 = (const float*)d_in[19];
  p.gdw = (const float*)d_in[20]; p.gdb = (const float*)d_in[21];
  p.gs2 = (const float*)d_in[22]; p.gb2 = (const float*)d_in[23];
  p.gm2 = (const float*)d_in[24]; p.gv2 = (const float*)d_in[25];
  p.fc1w = (const float*)d_in[27];
  p.wtE = wtE; p.wtG = wtG; p.f1T = f1T; p.sco = sco;

  const float* gamma = (const float*)d_in[26];
  const float* fc1b  = (const float*)d_in[28];
  const float* fc2w  = (const float*)d_in[29];
  const float* fc2b  = (const float*)d_in[30];

  prep_kernel<<<545, 256, 0, stream>>>(p);
  encoder_kernel<64, 64, 64, false><<<BB * 5, 256, 0, stream>>>(x_eeg, wtE, sco, encE);
  encoder_kernel<32, 18, 32, true ><<<BB * 5, 256, 0, stream>>>(x_geo, wtG, sco + 64, modG);
  scan_kernel<<<128, 256, 0, stream>>>(encE, modG, gamma, dpb);
  head_kernel<<<BB / 16, 256, 0, stream>>>(dpb, f1T, fc1b, fc2w, fc2b, (float*)d_out);
}
